// PerformerMultiHeadAttention_79860621902510
// MI455X (gfx1250) — matrix operand order repaired and checked by reading, not compiled, not run on hardware
//
#include <hip/hip_runtime.h>
#include <math.h>

// Performer multi-head attention, MI455X (gfx1250), bf16 WMMA pipeline.
// All GEMMs via v_wmma_f32_16x16x32_bf16 (f32 accumulate). Every WMMA operand
// is a contiguous 32 B/lane load (2 x b128); strided operands live in
// fragment-tiled layouts ([tile][lane][16 bf16]) packed once (weights, proj)
// or written for free by the producer's epilogue (Pv, Kf, KVZ) using closed-
// form index math + 16 B v8bf scatter stores. Long-K loops are register
// double-buffered so fragment loads overlap the WMMA group of the previous
// K-step. Stage 4 stages its (b,h) KVZ slice into LDS once per block with
// global_load_async_to_lds_b128 (ASYNCcnt) and reads B fragments from LDS.

#define DMODEL 1024
#define NHEAD  16
#define DK     64
#define MFEAT  256
#define BATCH  2
#define SEQ    4096
#define BLROWS (BATCH * SEQ)   // 8192
#define NDCOL  80              // DK + 16 pad; column 64 carries Z / den
#define EPS    1e-6f

typedef __bf16 bf16;
typedef __attribute__((ext_vector_type(16))) __bf16 v16bf;
typedef __attribute__((ext_vector_type(8)))  __bf16 v8bf;
typedef __attribute__((ext_vector_type(8)))  float  v8f;
typedef __attribute__((ext_vector_type(4)))  float  v4f;

static __device__ inline bf16 f2bf(float f) { return (bf16)f; }

static __device__ inline v8f vzero8() {
  v8f z = {0.f, 0.f, 0.f, 0.f, 0.f, 0.f, 0.f, 0.f};
  return z;
}

static __device__ inline v8f wmma_bf16(v16bf a, v16bf b, v8f c) {
  return __builtin_amdgcn_wmma_f32_16x16x32_bf16(false, a, false, b, (short)0, c,
                                                 false, false);
}

// ---------------- fragment-tiled index math (pack kernels only) -------------
// B fragment (32x16): lane = 16*hi | (n&15); element e <-> k = 16*hi + e
static __device__ inline long btile_idx(int NT, int k, int n) {
  int kt = k >> 5, nt = n >> 4;
  int kl = k & 31;
  int hi = kl >> 4;
  int e  = kl & 15;
  int lane = (hi << 4) | (n & 15);
  return ((((long)kt * NT + nt) * 32 + lane) << 4) + e;
}

// ---------------- fast fragment loaders (2 x b128 per lane) -----------------
static __device__ inline v16bf load_frag_tiled(const bf16* tile) {
  const int lane = threadIdx.x & 31;
  const bf16* r = tile + (lane << 4);
  v8bf lo = *(const v8bf*)(r);
  v8bf hh = *(const v8bf*)(r + 8);
  v16bf f;
#pragma unroll
  for (int i = 0; i < 8; ++i) { f[i] = lo[i]; f[8 + i] = hh[i]; }
  return f;
}

// A from row-major bf16 with contiguous K
static __device__ inline v16bf load_a_bf_fast(const bf16* p, long srow) {
  const int lane = threadIdx.x & 31;
  const int row = lane & 15, hi = lane >> 4;
  const bf16* r = p + (long)row * srow + 16 * hi;
  v8bf lo = *(const v8bf*)(r);        // k = 16hi .. 16hi+7
  v8bf hh = *(const v8bf*)(r + 8);    // k = 16hi+8 .. +7
  v16bf a;
#pragma unroll
  for (int i = 0; i < 8; ++i) { a[i] = lo[i]; a[8 + i] = hh[i]; }
  return a;
}

// A from row-major fp32 with contiguous K (4 x b128 + cvt_pk_bf16)
static __device__ inline v16bf load_a_f32_fast(const float* p, long srow) {
  const int lane = threadIdx.x & 31;
  const int row = lane & 15, hi = lane >> 4;
  const float* r = p + (long)row * srow + 16 * hi;
  v4f x0 = *(const v4f*)(r);
  v4f x1 = *(const v4f*)(r + 4);
  v4f x2 = *(const v4f*)(r + 8);
  v4f x3 = *(const v4f*)(r + 12);
  v16bf a;
#pragma unroll
  for (int i = 0; i < 4; ++i) {
    a[i]      = f2bf(x0[i]);
    a[4 + i]  = f2bf(x1[i]);
    a[8 + i]  = f2bf(x2[i]);
    a[12 + i] = f2bf(x3[i]);
  }
  return a;
}

// ---- stage 0: pack weights / proj into fragment-tiled bf16 -----------------
__global__ __launch_bounds__(256) void k_pack_w(const float* __restrict__ s,
                                                bf16* __restrict__ d, int n) {
  int i = blockIdx.x * 256 + threadIdx.x;
  if (i < n) {
    int k = i >> 10, c = i & 1023;              // W is K=1024 x N=1024 row-major
    d[btile_idx(64, k, c)] = f2bf(s[i]);
  }
}

__global__ __launch_bounds__(256) void k_pack_proj(const float* __restrict__ s,
                                                   bf16* __restrict__ d) {
  int i = blockIdx.x * 256 + threadIdx.x;      // H*M*DK = 262144 elements
  int h = i >> 14, rem = i & 16383;
  int m = rem >> 6, dd = rem & 63;             // proj[h][m][d]
  d[(long)h * 16384 + btile_idx(16, dd, m)] = f2bf(s[i]);
}

// ---- stage 1: P = X @ W + b  (wave tile 16x64, double-buffered K loop) -----
// vmode=0: row-major P (Pq/Pk). vmode=1: B-tiled per (b,h) (Pv).
__global__ __launch_bounds__(256) void k_proj(const float* __restrict__ X,
                                              const bf16* __restrict__ WT,
                                              const float* __restrict__ bias,
                                              bf16* __restrict__ P, int vmode) {
  const int wave = blockIdx.x * 8 + (threadIdx.x >> 5);
  const int cg = wave & 15;
  const int rt = wave >> 4;
  const long row0 = (long)rt * 16;
  const int col0 = cg * 64;
  v8f acc0 = vzero8(), acc1 = vzero8(), acc2 = vzero8(), acc3 = vzero8();
  const float* Xr = X + row0 * DMODEL;
  const bf16* wt = WT + ((long)(col0 >> 4) << 9);   // tile (kt=0, nt0) * 512
  const int wstep = 64 * 512;                       // next kt row of tiles
  v16bf a  = load_a_f32_fast(Xr, DMODEL);
  v16bf b0 = load_frag_tiled(wt);
  v16bf b1 = load_frag_tiled(wt + 512);
  v16bf b2 = load_frag_tiled(wt + 1024);
  v16bf b3 = load_frag_tiled(wt + 1536);
  for (int kk = 32; kk < DMODEL; kk += 32) {
    wt += wstep;
    __builtin_prefetch(Xr + kk + 96, 0, 1);
    v16bf an = load_a_f32_fast(Xr + kk, DMODEL);
    v16bf c0 = load_frag_tiled(wt);
    v16bf c1 = load_frag_tiled(wt + 512);
    v16bf c2 = load_frag_tiled(wt + 1024);
    v16bf c3 = load_frag_tiled(wt + 1536);
    acc0 = wmma_bf16(a, b0, acc0);
    acc1 = wmma_bf16(a, b1, acc1);
    acc2 = wmma_bf16(a, b2, acc2);
    acc3 = wmma_bf16(a, b3, acc3);
    a = an; b0 = c0; b1 = c1; b2 = c2; b3 = c3;
  }
  acc0 = wmma_bf16(a, b0, acc0);
  acc1 = wmma_bf16(a, b1, acc1);
  acc2 = wmma_bf16(a, b2, acc2);
  acc3 = wmma_bf16(a, b3, acc3);

  const int lane = threadIdx.x & 31;
  const int n = lane & 15, hi = lane >> 4;
  v8f accs[4] = {acc0, acc1, acc2, acc3};
  if (!vmode) {
#pragma unroll
    for (int t = 0; t < 4; ++t) {
      const int col = col0 + 16 * t + n;
      const float bv = bias[col];
#pragma unroll
      for (int r = 0; r < 8; ++r)
        P[(row0 + r + 8 * hi) * DMODEL + col] = f2bf(accs[t][r] + bv);
    }
  } else {
    // Pv -> per-(b,h) B-tiled. Closed form: h = cg; k = l = row local to slice;
    // kt = (row0&4095)>>5, hi_b = (row0>>4)&1, e = r + 8*hi (contiguous in r).
    const int b   = (int)(row0 >> 12);
    const int hib = (int)((row0 >> 4) & 1);
    const int kt  = (int)((row0 & 4095) >> 5);
    bf16* base = P + ((long)(b * 16 + cg) << 18);   // * 262144
#pragma unroll
    for (int t = 0; t < 4; ++t) {
      const float bv = bias[col0 + 16 * t + n];
      bf16* tb = base + ((((kt * 4 + t) * 32 + (hib << 4) + n) << 4) + 8 * hi);
      v8bf o;
#pragma unroll
      for (int r = 0; r < 8; ++r) o[r] = f2bf(accs[t][r] + bv);
      *(v8bf*)tb = o;                               // single 16 B store
    }
  }
}

// ---- stage 2: random features, fused exp epilogue --------------------------
__global__ __launch_bounds__(256) void k_rf(const bf16* __restrict__ Pq,
                                            const bf16* __restrict__ Pk,
                                            const bf16* __restrict__ projT,
                                            bf16* __restrict__ Qf,
                                            bf16* __restrict__ KfT) {
  const int wave = blockIdx.x * 8 + (threadIdx.x >> 5);
  const int mg  = wave & 3;
  const int lt  = (wave >> 2) & 255;
  const int bh  = (wave >> 10) & 31;
  const int sel = wave >> 15;                  // 0 = Q, 1 = K
  const int b = bh >> 4, h = bh & 15;
  const bf16* P = sel ? Pk : Pq;
  const float outscale = sel ? 1.0f : 0.0625f; // 1/sqrt(M)
  const bf16* pa = P + ((long)b * SEQ + (long)lt * 16) * DMODEL + (long)h * DK;
  const bf16* pb = projT + ((long)h << 14) + ((long)(mg * 4) << 9);
  v16bf a0 = load_a_bf_fast(pa, DMODEL);
  v16bf a1 = load_a_bf_fast(pa + 32, DMODEL);
  v8f acc[4] = {vzero8(), vzero8(), vzero8(), vzero8()};
#pragma unroll
  for (int t = 0; t < 4; ++t) {
    v16bf b0 = load_frag_tiled(pb + t * 512);          // kk = 0 tiles
    v16bf b1 = load_frag_tiled(pb + 8192 + t * 512);   // kk = 32 tiles
    acc[t] = wmma_bf16(a0, b0, acc[t]);
    acc[t] = wmma_bf16(a1, b1, acc[t]);
  }
  const int lane = threadIdx.x & 31;
  const int n = lane & 15, hi = lane >> 4;
  if (!sel) {
#pragma unroll
    for (int t = 0; t < 4; ++t) {
      const int mf = mg * 64 + 16 * t + n;
#pragma unroll
      for (int r = 0; r < 8; ++r) {
        const int l = lt * 16 + r + 8 * hi;
        Qf[((long)bh * SEQ + l) * MFEAT + mf] =
            f2bf(__expf(acc[t][r] * 0.125f) * outscale);
      }
    }
  } else {
    // Kf -> per-(b,h) A-tiled (rows=MFEAT, K=SEQ). Closed form:
    // kt = (lt>>1)+t, mt = mg*4, lane' = n + 16*(lt&1), e = r + 8*hi.
    bf16* base = KfT + ((long)bh << 20) +
                 (((((lt >> 1) * 16 + mg * 4) * 32 + n + ((lt & 1) << 4)) << 4) +
                  8 * hi);
#pragma unroll
    for (int t = 0; t < 4; ++t) {
      v8bf o;
#pragma unroll
      for (int r = 0; r < 8; ++r) o[r] = f2bf(__expf(acc[t][r] * 0.125f));
      *(v8bf*)(base + ((long)t << 13)) = o;   // t advances kt: +16*32*16 elems
    }
  }
}

// ---- stage 3: KVZ = Kf^T @ [V_head | 1 | 0]  (M x 80 per (b,h)) ------------
__global__ __launch_bounds__(256) void k_kvz(const bf16* __restrict__ KfT,
                                             const bf16* __restrict__ PvT,
                                             bf16* __restrict__ KVZT) {
  const int wave = blockIdx.x * 8 + (threadIdx.x >> 5);
  const int nt = wave % 5;
  const int mt = (wave / 5) & 15;
  const int bh = wave / 80;
  const int lane = threadIdx.x & 31;
  const int nn = lane & 15, hi = lane >> 4;
  v8f acc = vzero8();
  const bf16* at = KfT + ((long)bh << 20) + ((long)mt << 9);
  const int astep = 16 * 512;
  if (nt < 4) {  // wave-uniform branch: EXEC stays all-ones for WMMA
    const bf16* bt = PvT + ((long)bh << 18) + ((long)nt << 9);
    const int bstep = 4 * 512;
    v16bf a = load_frag_tiled(at);
    v16bf b = load_frag_tiled(bt);
    for (int kk = 32; kk < SEQ; kk += 32) {
      at += astep; bt += bstep;
      __builtin_prefetch(at + 2 * astep, 0, 1);
      __builtin_prefetch(bt + 2 * bstep, 0, 1);
      v16bf an = load_frag_tiled(at);
      v16bf bn = load_frag_tiled(bt);
      acc = wmma_bf16(a, b, acc);
      a = an; b = bn;
    }
    acc = wmma_bf16(a, b, acc);
  } else {
    // constant B fragment: global col 64 all-ones (-> Z), cols 65..79 zero
    bf16 hv = f2bf(nn == 0 ? 1.0f : 0.0f);
    v16bf bb;
#pragma unroll
    for (int i = 0; i < 16; ++i) bb[i] = hv;
    v16bf a = load_frag_tiled(at);
    for (int kk = 32; kk < SEQ; kk += 32) {
      at += astep;
      __builtin_prefetch(at + 2 * astep, 0, 1);
      v16bf an = load_frag_tiled(at);
      acc = wmma_bf16(a, bb, acc);
      a = an;
    }
    acc = wmma_bf16(a, bb, acc);
  }
  // KVZ -> per-(b,h) B-tiled. Closed form: kt = mt>>1, hi_b = mt&1,
  // lane' = 16*hi_b + nn, e = r + 8*hi (contiguous in r).
  bf16* dst = KVZT + (long)bh * 20480 +
              (((((mt >> 1) * 5 + nt) * 32 + ((mt & 1) << 4) + nn) << 4) + 8 * hi);
  v8bf o;
#pragma unroll
  for (int r = 0; r < 8; ++r) o[r] = f2bf(acc[r]);
  *(v8bf*)dst = o;
}

// ---- stage 4: Att_head = (Qf @ KVZ)[:, :64] / (col64 + eps) ----------------
// Per-block: async-stage the whole 40 KB KVZ (b,h) slice into LDS (ASYNCcnt),
// then all 5 waves read B fragments from LDS; epilogue divides via LDS strip.
__global__ __launch_bounds__(160) void k_att(const bf16* __restrict__ Qf,
                                             const bf16* __restrict__ KVZT,
                                             bf16* __restrict__ Att) {
  __shared__ __align__(16) bf16 sB[20480];      // 8*5 tiles * 512 = 40 KB
  __shared__ float s[16 * NDCOL];
  const int bh = blockIdx.x >> 8;
  const int lt = blockIdx.x & 255;
  const int b = bh >> 4, h = bh & 15;
  const int wv = threadIdx.x >> 5;              // nt = 0..4
  const int lane = threadIdx.x & 31;

  // cooperative async copy: 2560 x b128, 16 per thread, no divergence
  const unsigned lbase = (unsigned)(size_t)(&sB[0]);
  const unsigned long long gbase =
      (unsigned long long)(size_t)(KVZT + (long)bh * 20480);
  for (int i = threadIdx.x; i < 2560; i += 160) {
    unsigned loff = lbase + ((unsigned)i << 4);
    unsigned long long g = gbase + ((unsigned long long)i << 4);
    asm volatile("global_load_async_to_lds_b128 %0, %1, off"
                 :: "v"(loff), "v"(g) : "memory");
  }
  asm volatile("s_wait_asynccnt 0x0" ::: "memory");
  __syncthreads();

  const bf16* arow = Qf + ((long)bh * SEQ + (long)lt * 16) * MFEAT;
  v16bf afr[8];
#pragma unroll
  for (int i = 0; i < 8; ++i) afr[i] = load_a_bf_fast(arow + 32 * i, MFEAT);
  v8f acc = vzero8();
#pragma unroll
  for (int i = 0; i < 8; ++i) {
    const bf16* r = &sB[((i * 5 + wv) << 9) + (lane << 4)];
    v8bf lo = *(const v8bf*)(r);
    v8bf hh = *(const v8bf*)(r + 8);
    v16bf bb;
#pragma unroll
    for (int j = 0; j < 8; ++j) { bb[j] = lo[j]; bb[8 + j] = hh[j]; }
    acc = wmma_bf16(afr[i], bb, acc);
  }
  const int n = lane & 15, hi = lane >> 4;
#pragma unroll
  for (int r = 0; r < 8; ++r) s[(r + 8 * hi) * NDCOL + wv * 16 + n] = acc[r];
  __syncthreads();
  for (int idx = threadIdx.x; idx < 16 * DK; idx += 160) {
    int m = idx >> 6, d = idx & 63;
    float num = s[m * NDCOL + d];
    float den = s[m * NDCOL + DK] + EPS;
    Att[((long)b * SEQ + lt * 16 + m) * DMODEL + h * DK + d] = f2bf(num / den);
  }
}

// ---- stage 5: Out = Att @ Wo + bo  (fp32 out, double-buffered K loop) ------
__global__ __launch_bounds__(256) void k_out(const bf16* __restrict__ Att,
                                             const bf16* __restrict__ WoT,
                                             const float* __restrict__ bo,
                                             float* __restrict__ Out) {
  const int wave = blockIdx.x * 8 + (threadIdx.x >> 5);
  const int cg = wave & 15;
  const int rt = wave >> 4;
  const long row0 = (long)rt * 16;
  const int col0 = cg * 64;
  v8f acc0 = vzero8(), acc1 = vzero8(), acc2 = vzero8(), acc3 = vzero8();
  const bf16* Ar = Att + row0 * DMODEL;
  const bf16* wt = WoT + ((long)(col0 >> 4) << 9);
  const int wstep = 64 * 512;
  v16bf a  = load_a_bf_fast(Ar, DMODEL);
  v16bf b0 = load_frag_tiled(wt);
  v16bf b1 = load_frag_tiled(wt + 512);
  v16bf b2 = load_frag_tiled(wt + 1024);
  v16bf b3 = load_frag_tiled(wt + 1536);
  for (int kk = 32; kk < DMODEL; kk += 32) {
    wt += wstep;
    __builtin_prefetch(Ar + kk + 96, 0, 1);
    v16bf an = load_a_bf_fast(Ar + kk, DMODEL);
    v16bf c0 = load_frag_tiled(wt);
    v16bf c1 = load_frag_tiled(wt + 512);
    v16bf c2 = load_frag_tiled(wt + 1024);
    v16bf c3 = load_frag_tiled(wt + 1536);
    acc0 = wmma_bf16(a, b0, acc0);
    acc1 = wmma_bf16(a, b1, acc1);
    acc2 = wmma_bf16(a, b2, acc2);
    acc3 = wmma_bf16(a, b3, acc3);
    a = an; b0 = c0; b1 = c1; b2 = c2; b3 = c3;
  }
  acc0 = wmma_bf16(a, b0, acc0);
  acc1 = wmma_bf16(a, b1, acc1);
  acc2 = wmma_bf16(a, b2, acc2);
  acc3 = wmma_bf16(a, b3, acc3);

  const int lane = threadIdx.x & 31;
  const int n = lane & 15, hi = lane >> 4;
  v8f accs[4] = {acc0, acc1, acc2, acc3};
#pragma unroll
  for (int t = 0; t < 4; ++t) {
    float bv = bo[col0 + 16 * t + n];
#pragma unroll
    for (int r = 0; r < 8; ++r)
      Out[(row0 + r + 8 * hi) * DMODEL + col0 + 16 * t + n] = accs[t][r] + bv;
  }
}

extern "C" void kernel_launch(void* const* d_in, const int* in_sizes, int n_in,
                              void* d_out, int out_size, void* d_ws, size_t ws_size,
                              hipStream_t stream) {
  (void)in_sizes; (void)n_in; (void)out_size; (void)ws_size;
  const float* Q    = (const float*)d_in[0];
  const float* K    = (const float*)d_in[1];
  const float* V    = (const float*)d_in[2];
  const float* Wq   = (const float*)d_in[3];
  const float* bq   = (const float*)d_in[4];
  const float* Wk   = (const float*)d_in[5];
  const float* bk   = (const float*)d_in[6];
  const float* Wv   = (const float*)d_in[7];
  const float* bv   = (const float*)d_in[8];
  const float* Wo   = (const float*)d_in[9];
  const float* bo   = (const float*)d_in[10];
  const float* proj = (const float*)d_in[11];
  float* Out = (float*)d_out;

  char* w = (char*)d_ws;
  const size_t WEL  = (size_t)DMODEL * DMODEL;               // 1,048,576
  const size_t PEL  = (size_t)BLROWS * DMODEL;               // 8,388,608
  const size_t FEL  = (size_t)BATCH * NHEAD * SEQ * MFEAT;   // 33,554,432
  const size_t PREL = (size_t)NHEAD * 16384;                 // projT
  const size_t KVEL = (size_t)32 * 20480;                    // KVZT
  size_t off = 0;
  bf16* WqT   = (bf16*)(w + off); off += 2 * WEL;
  bf16* WkT   = (bf16*)(w + off); off += 2 * WEL;
  bf16* WvT   = (bf16*)(w + off); off += 2 * WEL;
  bf16* WoT   = (bf16*)(w + off); off += 2 * WEL;
  bf16* projT = (bf16*)(w + off); off += 2 * PREL;
  bf16* Pq    = (bf16*)(w + off); off += 2 * PEL;
  bf16* Pk    = (bf16*)(w + off); off += 2 * PEL;
  bf16* PvT   = (bf16*)(w + off); off += 2 * PEL;
  bf16* Qf    = (bf16*)(w + off); off += 2 * FEL;
  bf16* KfT   = (bf16*)(w + off); off += 2 * FEL;
  bf16* KVZT  = (bf16*)(w + off); off += 2 * KVEL;
  bf16* Att   = (bf16*)(w + off); off += 2 * PEL;

  const int wB = (int)((WEL + 255) / 256);
  k_pack_w<<<wB, 256, 0, stream>>>(Wq, WqT, (int)WEL);
  k_pack_w<<<wB, 256, 0, stream>>>(Wk, WkT, (int)WEL);
  k_pack_w<<<wB, 256, 0, stream>>>(Wv, WvT, (int)WEL);
  k_pack_w<<<wB, 256, 0, stream>>>(Wo, WoT, (int)WEL);
  k_pack_proj<<<(int)(PREL / 256), 256, 0, stream>>>(proj, projT);

  k_proj<<<1024, 256, 0, stream>>>(Q, WqT, bq, Pq, 0);
  k_proj<<<1024, 256, 0, stream>>>(K, WkT, bk, Pk, 0);
  k_proj<<<1024, 256, 0, stream>>>(V, WvT, bv, PvT, 1);

  k_rf<<<8192, 256, 0, stream>>>(Pq, Pk, projT, Qf, KfT);
  k_kvz<<<320, 256, 0, stream>>>(KfT, PvT, KVZT);
  k_att<<<8192, 160, 0, stream>>>(Qf, KVZT, Att);
  k_out<<<1024, 256, 0, stream>>>(Att, WoT, bo, Out);
}
